// GAT_25993142075919
// MI455X (gfx1250) — matrix operand order repaired and checked by reading, not compiled, not run on hardware
//
#include <hip/hip_runtime.h>
#include <hip/hip_bf16.h>
#include <stdint.h>

typedef __attribute__((ext_vector_type(16))) _Float16 v16h;
typedef __attribute__((ext_vector_type(8)))  float    v8f;
typedef int v4i_gat __attribute__((vector_size(16)));
typedef __attribute__((address_space(1))) v4i_gat* gptr_v4i;
typedef __attribute__((address_space(3))) v4i_gat* lptr_v4i;

#define GAT_ALPHA 0.2f
#define NTOT 8192

#if defined(__has_builtin)
#  if __has_builtin(__builtin_amdgcn_global_load_async_to_lds_b128)
#    define HAVE_ASYNC_LDS 1
#  endif
#  if __has_builtin(__builtin_amdgcn_s_wait_asynccnt)
#    define HAVE_WAIT_ASYNC 1
#  endif
#endif

__device__ __forceinline__ float leaky(float x){ return x > 0.f ? x : GAT_ALPHA * x; }

// 16B global -> LDS copy, async on gfx1250 (ASYNCcnt path), plain fallback.
__device__ __forceinline__ void copy16_g2l(const void* g, void* l) {
#ifdef HAVE_ASYNC_LDS
  __builtin_amdgcn_global_load_async_to_lds_b128(
      (gptr_v4i)(v4i_gat*)g, (lptr_v4i)(v4i_gat*)l, 0, 0);
#else
  *(float4*)l = *(const float4*)g;
#endif
}

__device__ __forceinline__ void wait_async0() {
#ifdef HAVE_WAIT_ASYNC
  __builtin_amdgcn_s_wait_asynccnt(0);
#else
  asm volatile("s_wait_asynccnt 0x0" ::: "memory");
#endif
}

// ---------------------------------------------------------------------------
// Pack adj (int32 0/1) -> bitmask, 1 bit per edge. 256MB -> 8MB (L2-resident).
// ---------------------------------------------------------------------------
__global__ void k_pack_adj(const int4* __restrict__ adj4, uint32_t* __restrict__ mask,
                           int nwords) {
  int w = blockIdx.x * blockDim.x + threadIdx.x;
  if (w >= nwords) return;
  const int4* p = adj4 + (size_t)w * 8;
  uint32_t bits = 0u;
#pragma unroll
  for (int q = 0; q < 8; ++q) {
    int4 v = p[q];
    bits |= (v.x > 0 ? 1u : 0u) << (q * 4 + 0);
    bits |= (v.y > 0 ? 1u : 0u) << (q * 4 + 1);
    bits |= (v.z > 0 ? 1u : 0u) << (q * 4 + 2);
    bits |= (v.w > 0 ? 1u : 0u) << (q * 4 + 3);
  }
  mask[w] = bits;
}

// f2 stored permuted within each 32-block to match the A-fragment K order:
// positions [0..7]=K0..7, [8..15]=K16..23, [16..23]=K8..15, [24..31]=K24..31.
__device__ __forceinline__ int f2pos(int j) {
  int q = (j >> 3) & 3;
  int qp = (q == 1) ? 2 : (q == 2) ? 1 : q;
  return j - q * 8 + qp * 8;
}

// ---------------------------------------------------------------------------
// Layer-1 prep: h = x@W per head, f1=h@a1, f2(permuted)=h@a2,
// hT[head][16][N] f16 with cols 0..7 = h, col 15 = 1.0 (free row-sum column).
// ---------------------------------------------------------------------------
__global__ void k_prep1(const float* __restrict__ x, const float* __restrict__ W_heads,
                        const float* __restrict__ a_heads, float* __restrict__ f1,
                        float* __restrict__ f2A, _Float16* __restrict__ hT, int N) {
  __shared__ float sW[4 * 29 * 8];
  __shared__ float sa[4 * 16];
  for (int i = threadIdx.x; i < 4 * 29 * 8; i += 256) sW[i] = W_heads[i];
  for (int i = threadIdx.x; i < 64; i += 256) sa[i] = a_heads[i];
  __syncthreads();
  int j = blockIdx.x * 256 + threadIdx.x;
  if (j >= N) return;
  float xr[29];
#pragma unroll
  for (int f = 0; f < 29; ++f) xr[f] = x[(size_t)j * 29 + f];
  for (int head = 0; head < 4; ++head) {
    float h[8];
#pragma unroll
    for (int c = 0; c < 8; ++c) h[c] = 0.f;
#pragma unroll
    for (int f = 0; f < 29; ++f) {
      float xv = xr[f];
      const float* wr = &sW[head * 232 + f * 8];
#pragma unroll
      for (int c = 0; c < 8; ++c) h[c] = __builtin_fmaf(xv, wr[c], h[c]);
    }
    float s1 = 0.f, s2 = 0.f;
#pragma unroll
    for (int c = 0; c < 8; ++c) {
      s1 = __builtin_fmaf(h[c], sa[head * 16 + c], s1);
      s2 = __builtin_fmaf(h[c], sa[head * 16 + 8 + c], s2);
    }
    f1[head * N + j] = s1;
    f2A[head * N + f2pos(j)] = s2;
    _Float16* hTh = hT + (size_t)head * 16 * N;
#pragma unroll
    for (int c = 0; c < 16; ++c) {
      _Float16 v = (c < 8) ? (_Float16)h[c] : (c == 15 ? (_Float16)1.0f : (_Float16)0.0f);
      hTh[(size_t)c * N + j] = v;
    }
  }
}

__global__ void k_rmax(const float* __restrict__ v, float* __restrict__ out, int N) {
  __shared__ float red[256];
  const float* p = v + (size_t)blockIdx.x * N;
  float m = -3.0e38f;
  for (int i = threadIdx.x; i < N; i += 256) m = fmaxf(m, p[i]);
  red[threadIdx.x] = m;
  __syncthreads();
  for (int s = 128; s > 0; s >>= 1) {
    if (threadIdx.x < s) red[threadIdx.x] = fmaxf(red[threadIdx.x], red[threadIdx.x + s]);
    __syncthreads();
  }
  if (threadIdx.x == 0) out[blockIdx.x] = red[0];
}

// ---------------------------------------------------------------------------
// Fused masked-softmax + attn@h. Block = 8 waves, all on the same head;
// each wave owns one 16-row block. hT / f2 super-chunks (16x1024 f16 + 1024
// f32 = 36KB) are staged in LDS with async global->LDS (ASYNCcnt) so the 8
// waves share one copy instead of 8 L2 streams. A-fragment scores built
// in-register in exact CDNA5 order; Z row-sum arrives in C column 15.
// ---------------------------------------------------------------------------
#define JCHUNK 1024
__global__ void k_attn(const uint32_t* __restrict__ mask, const float* __restrict__ f1,
                       const float* __restrict__ f2A, const float* __restrict__ f2max,
                       const _Float16* __restrict__ hT, float* __restrict__ out,
                       int N, int groupsPerHead, int out_ld, int fvalid, int colstride) {
  __shared__ __align__(16) _Float16 sB[16 * JCHUNK];   // [c][JCHUNK]  32KB
  __shared__ __align__(16) float    sF2[JCHUNK];       // permuted      4KB

  const int lane = threadIdx.x & 31;
  const int wv   = threadIdx.x >> 5;                    // wave in block (0..7)
  const int head = blockIdx.x / groupsPerHead;
  const int grp  = blockIdx.x % groupsPerHead;
  const int i0   = (grp * 8 + wv) * 16;
  const int half = lane >> 4;      // 0: K 0..7,16..23   1: K 8..15,24..31
  const int c    = lane & 15;      // A row (M) for score gen; B/D column

  const int row = i0 + c;
  const float f1v  = f1[head * N + row];
  const float stab = leaky(f1v + f2max[head]);   // e_ij <= stab => w <= 1 (f16-safe)
  const uint32_t* mrow = mask + (size_t)row * (N >> 5);
  const float*    f2h  = f2A + head * N;
  const _Float16* hTh  = hT + (size_t)head * 16 * N;

  v8f acc = {0.f, 0.f, 0.f, 0.f, 0.f, 0.f, 0.f, 0.f};
  const int nsc = N / JCHUNK;                           // 8 super-chunks
  for (int sc = 0; sc < nsc; ++sc) {
    __syncthreads();                                    // prev chunk consumed
    // stage hT chunk: 16 rows x 1024 f16 = 2048 16B units, 8 per thread
#pragma unroll
    for (int q = 0; q < 8; ++q) {
      int u  = q * 256 + threadIdx.x;
      int cc = u >> 7;
      int jj = (u & 127) << 3;
      copy16_g2l(hTh + (size_t)cc * N + sc * JCHUNK + jj, &sB[cc * JCHUNK + jj]);
    }
    // stage f2 chunk: 1024 f32 = 256 16B units, 1 per thread
    copy16_g2l(f2h + sc * JCHUNK + threadIdx.x * 4, &sF2[threadIdx.x * 4]);
    wait_async0();
    __syncthreads();

#pragma unroll 2
    for (int tt = 0; tt < JCHUNK / 32; ++tt) {
      uint32_t mw = mrow[sc * (JCHUNK / 32) + tt];
      v16h B;
#pragma unroll
      for (int k = 0; k < 16; ++k) B[k] = sB[c * JCHUNK + tt * 32 + (k < 8 ? k : k + 8) + half * 8];
      float fv[16];
#pragma unroll
      for (int k = 0; k < 16; ++k) fv[k] = sF2[tt * 32 + half * 16 + k];
      v16h A;
#pragma unroll
      for (int k = 0; k < 16; ++k) {
        int K = (k < 8 ? k : k + 8) + half * 8;        // mask bit index
        float s = leaky(f1v + fv[k]);
        float w = __expf(s - stab);
        w = ((mw >> K) & 1u) ? w : 0.f;
        A[k] = (_Float16)w;
      }
      acc = __builtin_amdgcn_wmma_f32_16x16x32_f16(false, A, false, B,
                                                   (short)0, acc, false, false);
    }
  }

  // C layout: lane holds column c for rows M = v + half*8. Column 15 = Z.
#pragma unroll
  for (int v = 0; v < 8; ++v) {
    float z  = __shfl(acc[v], half ? 31 : 15, 32);
    float val = acc[v] * __builtin_amdgcn_rcpf(z);     // softmax normalize
    float e = val > 0.f ? val : (__expf(val) - 1.0f);  // ELU
    if (c < fvalid) {
      int M = v + half * 8;
      out[(size_t)(i0 + M) * out_ld + head * colstride + c] = e;
    }
  }
}

// ---------------------------------------------------------------------------
// Layer-2 prep: h2 = hcat@W_out (32->2), f1/f2 from a_out, hT2 with col15=1.
// ---------------------------------------------------------------------------
__global__ void k_prep2(const float* __restrict__ hcat, const float* __restrict__ W_out,
                        const float* __restrict__ a_out, float* __restrict__ f1b,
                        float* __restrict__ f2Ab, _Float16* __restrict__ hT2, int N) {
  int j = blockIdx.x * 256 + threadIdx.x;
  if (j >= N) return;
  float h0 = 0.f, h1 = 0.f;
#pragma unroll
  for (int f = 0; f < 32; ++f) {
    float hv = hcat[(size_t)j * 32 + f];
    h0 = __builtin_fmaf(hv, W_out[f * 2 + 0], h0);
    h1 = __builtin_fmaf(hv, W_out[f * 2 + 1], h1);
  }
  f1b[j] = h0 * a_out[0] + h1 * a_out[1];
  f2Ab[f2pos(j)] = h0 * a_out[2] + h1 * a_out[3];
#pragma unroll
  for (int cc = 0; cc < 16; ++cc) {
    _Float16 v = (cc == 0) ? (_Float16)h0
               : (cc == 1) ? (_Float16)h1
               : (cc == 15) ? (_Float16)1.0f : (_Float16)0.0f;
    hT2[(size_t)cc * N + j] = v;
  }
}

__global__ void k_logsoftmax(const float* __restrict__ h2o, float* __restrict__ out, int N) {
  int j = blockIdx.x * 256 + threadIdx.x;
  if (j >= N) return;
  float a = h2o[j * 2], b = h2o[j * 2 + 1];
  float m = fmaxf(a, b);
  float l = m + __logf(__expf(a - m) + __expf(b - m));
  out[j * 2] = a - l;
  out[j * 2 + 1] = b - l;
}

extern "C" void kernel_launch(void* const* d_in, const int* in_sizes, int n_in,
                              void* d_out, int out_size, void* d_ws, size_t ws_size,
                              hipStream_t stream) {
  (void)in_sizes; (void)n_in; (void)out_size; (void)ws_size;
  const float* x       = (const float*)d_in[0];
  const int*   adj     = (const int*)d_in[1];
  const float* W_heads = (const float*)d_in[2];
  const float* a_heads = (const float*)d_in[3];
  const float* W_out   = (const float*)d_in[4];
  const float* a_out   = (const float*)d_in[5];
  const int N = NTOT;

  char* ws = (char*)d_ws;
  size_t off = 0;
  auto carve = [&](size_t bytes) -> char* {
    char* p = ws + off;
    off = (off + bytes + 255) & ~(size_t)255;
    return p;
  };
  uint32_t* mask = (uint32_t*)carve((size_t)N * N / 8);       // 8 MB bitmask
  float*    f1   = (float*)carve((size_t)4 * N * 4);
  float*    f2A  = (float*)carve((size_t)4 * N * 4);
  _Float16* hT   = (_Float16*)carve((size_t)4 * 16 * N * 2);  // 1 MB
  float*    hcat = (float*)carve((size_t)N * 32 * 4);         // 1 MB
  float*    f2m  = (float*)carve(256);
  float*    f1b  = (float*)carve((size_t)N * 4);
  float*    f2Ab = (float*)carve((size_t)N * 4);
  _Float16* hT2  = (_Float16*)carve((size_t)16 * N * 2);
  float*    f2mb = (float*)carve(256);
  float*    h2o  = (float*)carve((size_t)N * 2 * 4);

  const int nwords = N * N / 32;
  k_pack_adj<<<(nwords + 255) / 256, 256, 0, stream>>>((const int4*)adj, mask, nwords);
  k_prep1<<<N / 256, 256, 0, stream>>>(x, W_heads, a_heads, f1, f2A, hT, N);
  k_rmax<<<4, 256, 0, stream>>>(f2A, f2m, N);
  // layer 1: 4 heads x 64 groups (8 waves x 16 rows each) = 256 blocks
  k_attn<<<4 * (N / 128), 256, 0, stream>>>(mask, f1, f2A, f2m, hT, hcat,
                                            N, N / 128, 32, 8, 8);
  k_prep2<<<N / 256, 256, 0, stream>>>(hcat, W_out, a_out, f1b, f2Ab, hT2, N);
  k_rmax<<<1, 256, 0, stream>>>(f2Ab, f2mb, N);
  k_attn<<<N / 128, 256, 0, stream>>>(mask, f1b, f2Ab, f2mb, hT2, h2o,
                                      N, N / 128, 2, 2, 0);
  k_logsoftmax<<<N / 256, 256, 0, stream>>>(h2o, (float*)d_out, N);
}
